// GraphAttentionLayer_23785528885547
// MI455X (gfx1250) — matrix operand order repaired and checked by reading, not compiled, not run on hardware
//
#include <hip/hip_runtime.h>
#include <hip/hip_bf16.h>

#define N_NODES 8192
#define IN_F    256
#define OUT_F   128
#define LRELU_ALPHA 0.2f

typedef __attribute__((ext_vector_type(16))) _Float16 v16h;
typedef __attribute__((ext_vector_type(8)))  float    v8f;

struct alignas(16) H8 { _Float16 s[8]; };
struct alignas(16) I4 { int      s[4]; };
struct alignas(16) F4 { float    s[4]; };

// ---------------- kernel 1a: h (f32) -> h16 (f16) ----------------
__global__ void k_cvt_h(const float* __restrict__ h, _Float16* __restrict__ h16, int n) {
    int i = blockIdx.x * blockDim.x + threadIdx.x;
    if (i < n) h16[i] = (_Float16)h[i];
}

// ---------------- kernel 1b: W (256x128 f32) -> Wt16 (128x256 f16, transposed) ----------------
__global__ void k_cvt_wt(const float* __restrict__ W, _Float16* __restrict__ wt16) {
    int i = blockIdx.x * blockDim.x + threadIdx.x;
    if (i < IN_F * OUT_F) {
        int k = i / OUT_F, n = i % OUT_F;
        wt16[n * IN_F + k] = (_Float16)W[i];
    }
}

// ---------------- kernel 2: Wh = h @ W via WMMA f16 ----------------
// One wave per 16x16 output tile. Writes Wh (f32, row-major) and WhT (f16, 128 x 8192).
__global__ __launch_bounds__(128) void k_gemm_wh(const _Float16* __restrict__ h16,
                                                 const _Float16* __restrict__ wt16,
                                                 float* __restrict__ wh,
                                                 _Float16* __restrict__ whT) {
    int tile = blockIdx.x * 4 + threadIdx.y;     // 4096 tiles = 512 row-tiles x 8 col-tiles
    int it = tile >> 3, nt = tile & 7;
    int i0 = it << 4, n0 = nt << 4;
    int lane = threadIdx.x;
    int half = lane >> 4, l15 = lane & 15;

    v8f z = {0.f,0.f,0.f,0.f,0.f,0.f,0.f,0.f};
    v8f acc = z;

    const _Float16* arow = h16  + (size_t)(i0 + l15) * IN_F;   // A: M = i0+l15 (per lane group)
    const _Float16* brow = wt16 + (size_t)(n0 + l15) * IN_F;   // B: N = n0+l15

#pragma unroll
    for (int kc = 0; kc < 8; ++kc) {
        int k0 = kc << 5;
        // A 16x32 f16 layout: lanes 0-15 hold K {0..7,16..23}; lanes 16-31 hold K {8..15,24..31}
        H8 a0 = *(const H8*)(arow + k0 + half * 8);
        H8 a1 = *(const H8*)(arow + k0 + 16 + half * 8);
        // B 32x16 f16 layout: lanes 0-15 hold K {0..7,16..23}; lanes 16-31 hold K {8..15,24..31}
        H8 b0 = *(const H8*)(brow + k0 + half * 8);
        H8 b1 = *(const H8*)(brow + k0 + 16 + half * 8);
        v16h A, B;
#pragma unroll
        for (int t = 0; t < 8; ++t) {
            A[t] = a0.s[t]; A[8 + t] = a1.s[t];
            B[t] = b0.s[t]; B[8 + t] = b1.s[t];
        }
        acc = __builtin_amdgcn_wmma_f32_16x16x32_f16(false, A, false, B, (short)0, acc,
                                                     false, false);
    }

    // C/D layout: lanes 0-15 VGPR r -> row r; lanes 16-31 VGPR r -> row 8+r; col = lane%16
    int rbase = i0 + half * 8;
    int col   = n0 + l15;
    H8 pk;
#pragma unroll
    for (int r = 0; r < 8; ++r) {
        wh[(size_t)(rbase + r) * OUT_F + col] = acc[r];
        pk.s[r] = (_Float16)acc[r];
    }
    *(H8*)(whT + (size_t)col * N_NODES + rbase) = pk;   // 16B contiguous transpose store
}

// ---------------- kernel 3: e1[i]=Wh[i,:]·a1, e2[i]=Wh[i,:]·a2 ----------------
__global__ void k_e12(const float* __restrict__ wh, const float* __restrict__ a,
                      float* __restrict__ e1, float* __restrict__ e2) {
    int i = blockIdx.x * blockDim.x + threadIdx.x;
    if (i >= N_NODES) return;
    const float* row = wh + (size_t)i * OUT_F;
    float s1 = 0.f, s2 = 0.f;
#pragma unroll 4
    for (int n = 0; n < OUT_F; ++n) {
        float v = row[n];
        s1 += v * a[n];
        s2 += v * a[OUT_F + n];
    }
    e1[i] = s1; e2[i] = s2;
}

// ---------------- kernel 3b: e2max = max_j e2[j] ----------------
__global__ void k_maxe2(const float* __restrict__ e2, float* __restrict__ e2max) {
    __shared__ float red[256];
    float m = -3.4e38f;
    for (int i = threadIdx.x; i < N_NODES; i += 256) m = fmaxf(m, e2[i]);
    red[threadIdx.x] = m;
    __syncthreads();
    for (int s = 128; s > 0; s >>= 1) {
        if (threadIdx.x < s) red[threadIdx.x] = fmaxf(red[threadIdx.x], red[threadIdx.x + s]);
        __syncthreads();
    }
    if (threadIdx.x == 0) *e2max = red[0];
}

// ---------------- kernel 4: fused mask + softmax + attention@Wh + ELU ----------------
// 256 blocks x 128 threads (4 waves). Block owns 32 rows (2 A-tiles); waves interleave
// 32-wide j chunks so all 4 waves touch the same WhT slice concurrently (WGP$ sharing).
// Each B tile load feeds 2 WMMAs (one per A row-tile): halves WhT L2 traffic vs 16-row blocks.
__global__ __launch_bounds__(128) void k_gat_main(const int* __restrict__ adj,
                                                  const _Float16* __restrict__ whT,
                                                  const float* __restrict__ e1,
                                                  const float* __restrict__ e2,
                                                  const float* __restrict__ e2max_p,
                                                  float* __restrict__ out) {
    __shared__ float s_acc[32 * OUT_F];
    __shared__ float s_rs[32];

    int tid  = threadIdx.x;
    int wave = tid >> 5, lane = tid & 31;
    int half = lane >> 4, l15 = lane & 15;
    int i0   = blockIdx.x << 5;          // 32 rows per block
    int irow0 = i0 + l15;                // A row-tile 0 row for this lane (both halves)
    int irow1 = i0 + 16 + l15;           // A row-tile 1 row

    float e2m = *e2max_p;
    float e1v0 = e1[irow0];
    float e1v1 = e1[irow1];
    float x0 = e1v0 + e2m;
    float x1 = e1v1 + e2m;
    float Mi0 = x0 > 0.f ? x0 : LRELU_ALPHA * x0;  // static upper bound on row max (LR monotone)
    float Mi1 = x1 > 0.f ? x1 : LRELU_ALPHA * x1;

    const int* arow0 = adj + (size_t)irow0 * N_NODES;
    const int* arow1 = adj + (size_t)irow1 * N_NODES;
    int kb0 = half * 8;                  // K base within chunk for this lane half

    v8f z = {0.f,0.f,0.f,0.f,0.f,0.f,0.f,0.f};
    v8f acc0[8] = {z, z, z, z, z, z, z, z};
    v8f acc1[8] = {z, z, z, z, z, z, z, z};
    float rs0 = 0.f, rs1 = 0.f;

    for (int c = wave; c < N_NODES / 32; c += 4) {   // interleaved chunk assignment
        int j0 = c << 5;
        __builtin_prefetch(arow0 + j0 + 128 + kb0, 0, 1);  // next chunk for this wave
        __builtin_prefetch(arow1 + j0 + 128 + kb0, 0, 1);

        I4 A00 = *(const I4*)(arow0 + j0 + kb0);
        I4 A01 = *(const I4*)(arow0 + j0 + kb0 + 4);
        I4 A02 = *(const I4*)(arow0 + j0 + kb0 + 16);
        I4 A03 = *(const I4*)(arow0 + j0 + kb0 + 20);
        I4 A10 = *(const I4*)(arow1 + j0 + kb0);
        I4 A11 = *(const I4*)(arow1 + j0 + kb0 + 4);
        I4 A12 = *(const I4*)(arow1 + j0 + kb0 + 16);
        I4 A13 = *(const I4*)(arow1 + j0 + kb0 + 20);
        F4 E0  = *(const F4*)(e2 + j0 + kb0);        // shared by both row-tiles
        F4 E1  = *(const F4*)(e2 + j0 + kb0 + 4);
        F4 E2  = *(const F4*)(e2 + j0 + kb0 + 16);
        F4 E3  = *(const F4*)(e2 + j0 + kb0 + 20);

        float p0[16], p1[16];
#pragma unroll
        for (int t = 0; t < 4; ++t) {
            float ea = E0.s[t], eb = E1.s[t], ec = E2.s[t], ed = E3.s[t];
            float xa0 = e1v0 + ea; xa0 = xa0 > 0.f ? xa0 : LRELU_ALPHA * xa0;
            float xb0 = e1v0 + eb; xb0 = xb0 > 0.f ? xb0 : LRELU_ALPHA * xb0;
            float xc0 = e1v0 + ec; xc0 = xc0 > 0.f ? xc0 : LRELU_ALPHA * xc0;
            float xd0 = e1v0 + ed; xd0 = xd0 > 0.f ? xd0 : LRELU_ALPHA * xd0;
            float xa1 = e1v1 + ea; xa1 = xa1 > 0.f ? xa1 : LRELU_ALPHA * xa1;
            float xb1 = e1v1 + eb; xb1 = xb1 > 0.f ? xb1 : LRELU_ALPHA * xb1;
            float xc1 = e1v1 + ec; xc1 = xc1 > 0.f ? xc1 : LRELU_ALPHA * xc1;
            float xd1 = e1v1 + ed; xd1 = xd1 > 0.f ? xd1 : LRELU_ALPHA * xd1;
            p0[t]      = (A00.s[t] > 0) ? __expf(xa0 - Mi0) : 0.f;
            p0[4 + t]  = (A01.s[t] > 0) ? __expf(xb0 - Mi0) : 0.f;
            p0[8 + t]  = (A02.s[t] > 0) ? __expf(xc0 - Mi0) : 0.f;
            p0[12 + t] = (A03.s[t] > 0) ? __expf(xd0 - Mi0) : 0.f;
            p1[t]      = (A10.s[t] > 0) ? __expf(xa1 - Mi1) : 0.f;
            p1[4 + t]  = (A11.s[t] > 0) ? __expf(xb1 - Mi1) : 0.f;
            p1[8 + t]  = (A12.s[t] > 0) ? __expf(xc1 - Mi1) : 0.f;
            p1[12 + t] = (A13.s[t] > 0) ? __expf(xd1 - Mi1) : 0.f;
        }

        v16h Pm0, Pm1;
#pragma unroll
        for (int t = 0; t < 16; ++t) {
            rs0 += p0[t]; Pm0[t] = (_Float16)p0[t];
            rs1 += p1[t]; Pm1[t] = (_Float16)p1[t];
        }

#pragma unroll
        for (int nt = 0; nt < 8; ++nt) {
            // B 32x16: lane column N = nt*16 + l15; K {8h..8h+7, 16+8h..16+8h+7} per half
            const _Float16* bp = whT + (size_t)(nt * 16 + l15) * N_NODES + j0 + half * 8;
            H8 b0 = *(const H8*)bp;
            H8 b1 = *(const H8*)(bp + 16);
            v16h B;
#pragma unroll
            for (int t = 0; t < 8; ++t) { B[t] = b0.s[t]; B[8 + t] = b1.s[t]; }
            acc0[nt] = __builtin_amdgcn_wmma_f32_16x16x32_f16(false, Pm0, false, B, (short)0,
                                                              acc0[nt], false, false);
            acc1[nt] = __builtin_amdgcn_wmma_f32_16x16x32_f16(false, Pm1, false, B, (short)0,
                                                              acc1[nt], false, false);
        }
    }

    // cross-wave reduction in LDS
    for (int t = tid; t < 32 * OUT_F; t += 128) s_acc[t] = 0.f;
    if (tid < 32) s_rs[tid] = 0.f;
    __syncthreads();

    atomicAdd(&s_rs[l15], rs0);          // lanes l and l+16 both hold row l partial sums
    atomicAdd(&s_rs[16 + l15], rs1);
#pragma unroll
    for (int nt = 0; nt < 8; ++nt) {
#pragma unroll
        for (int r = 0; r < 8; ++r) {
            atomicAdd(&s_acc[(half * 8 + r) * OUT_F + nt * 16 + l15], acc0[nt][r]);
            atomicAdd(&s_acc[(16 + half * 8 + r) * OUT_F + nt * 16 + l15], acc1[nt][r]);
        }
    }
    __syncthreads();

    for (int t = tid; t < 32 * OUT_F; t += 128) {
        int r = t >> 7, n = t & (OUT_F - 1);
        float l = fmaxf(s_rs[r], 1e-30f);
        float x = s_acc[t] / l;
        out[(size_t)(i0 + r) * OUT_F + n] = x > 0.f ? x : (__expf(x) - 1.f);
    }
}

// ---------------- launcher ----------------
extern "C" void kernel_launch(void* const* d_in, const int* in_sizes, int n_in,
                              void* d_out, int out_size, void* d_ws, size_t ws_size,
                              hipStream_t stream) {
    const float* h   = (const float*)d_in[0];
    const int*   adj = (const int*)  d_in[1];
    const float* W   = (const float*)d_in[2];
    const float* a   = (const float*)d_in[3];
    float*       out = (float*)d_out;

    char* ws = (char*)d_ws;
    // workspace layout (≈10.2 MB total)
    _Float16* h16   = (_Float16*)(ws);                       // 8192*256*2  = 4 MiB
    _Float16* wt16  = (_Float16*)(ws + 4194304);             // 128*256*2   = 64 KiB
    float*    wh    = (float*)   (ws + 4259840);             // 8192*128*4  = 4 MiB
    _Float16* whT   = (_Float16*)(ws + 8454144);             // 128*8192*2  = 2 MiB
    float*    e1    = (float*)   (ws + 10551296);            // 32 KiB
    float*    e2    = (float*)   (ws + 10584064);            // 32 KiB
    float*    e2max = (float*)   (ws + 10616832);            // 4 B

    // 1) precision conversion (+ W transpose)
    k_cvt_h<<<(N_NODES * IN_F + 255) / 256, 256, 0, stream>>>(h, h16, N_NODES * IN_F);
    k_cvt_wt<<<(IN_F * OUT_F + 255) / 256, 256, 0, stream>>>(W, wt16);

    // 2) Wh = h @ W   (WMMA f16, fp32 accumulate)
    k_gemm_wh<<<1024, dim3(32, 4), 0, stream>>>(h16, wt16, wh, whT);

    // 3) attention logits pieces
    k_e12<<<(N_NODES + 255) / 256, 256, 0, stream>>>(wh, a, e1, e2);
    k_maxe2<<<1, 256, 0, stream>>>(e2, e2max);

    // 4) fused masked softmax + attention@Wh + ELU (single pass over adj, 32 rows/block)
    k_gat_main<<<N_NODES / 32, 128, 0, stream>>>(adj, whT, e1, e2, e2max, out);
}
